// SelfAttent_89017492177017
// MI455X (gfx1250) — matrix operand order repaired and checked by reading, not compiled, not run on hardware
//
#include <hip/hip_runtime.h>

typedef __attribute__((ext_vector_type(16))) __bf16 v16bf;
typedef __attribute__((ext_vector_type(8)))  float  v8f;
typedef unsigned short u16;
typedef unsigned int   u32;
typedef unsigned long long u64;

#define H_   16
#define DQ_  64
#define B_   2
#define S_   2048
#define D_   1024
#define MTOT (B_ * S_)   // 4096 rows

// ---------------- TDM availability ----------------
#if defined(__has_builtin)
#  if __has_builtin(__builtin_amdgcn_tensor_load_to_lds) && \
      __has_builtin(__builtin_amdgcn_s_wait_tensorcnt)
#    define HAS_TDM 1
#  endif
#endif
#ifndef HAS_TDM
#  define HAS_TDM 0
#endif

// -------- helpers --------
__device__ __forceinline__ u16 f2bf(float f) {           // RNE fp32 -> bf16
  u32 u = __builtin_bit_cast(u32, f);
  u = (u + 0x7FFFu + ((u >> 16) & 1u)) >> 16;
  return (u16)u;
}

union Frag {                 // one WMMA 16-bit A/B operand (8 VGPRs)
  v16bf v;
  uint4 q[2];
};

__device__ __forceinline__ v8f wmma_bf16(v16bf a, v16bf b, v8f c) {
  // D = A(16x32 bf16) * B(32x16 bf16) + C(f32)
  return __builtin_amdgcn_wmma_f32_16x16x32_bf16(false, a, false, b,
                                                 (short)0, c, false, false);
}

#if HAS_TDM
typedef __attribute__((ext_vector_type(4))) unsigned int tdm_v4u;
typedef __attribute__((ext_vector_type(8))) int          tdm_v8i;
typedef __attribute__((ext_vector_type(4))) int          tdm_v4i;

// 2D TDM tile load: global (row-major, 2-byte elems, row length stride0)
// -> LDS at ldst, with LDS row padding (pad_amount+1 dwords every
// 2^(pad_interval+1) dwords) to realize bank-conflict-free strides.
__device__ __forceinline__ void tdm_load_2d(
    const u16* gsrc, const u16* ldst,
    u32 dim0, u32 dim1, u32 stride0,
    u32 tile0, u32 tile1, u32 pad_interval, u32 pad_amount)
{
  u64 ga = (u64)(uintptr_t)gsrc;                     // 57-bit global byte addr
  u32 lo = (u32)(uintptr_t)ldst;                     // flat[31:0] == LDS offset
  u32 w1 = (u32)__builtin_amdgcn_readfirstlane((int)lo);
  u32 w2 = (u32)__builtin_amdgcn_readfirstlane((int)(u32)ga);
  u32 w3 = (u32)__builtin_amdgcn_readfirstlane(
               (int)(((u32)(ga >> 32) & 0x1FFFFFFu) | (2u << 30))); // type=2
  tdm_v4u g0 = { 1u /*count=1*/, w1, w2, w3 };
  // group1: data_size=1 (2B), pad_enable, pad geometry; no multicast/barrier
  u32 p0 = (1u << 16) | (1u << 20) | (pad_interval << 22) | (pad_amount << 25);
  u32 p1 = (dim0 & 0xFFFFu) << 16;                   // tensor_dim0[15:0]
  u32 p2 = (dim0 >> 16) | ((dim1 & 0xFFFFu) << 16);  // dim0[31:16], dim1[15:0]
  u32 p3 = (dim1 >> 16) | (tile0 << 16);             // dim1[31:16], tile_dim0
  u32 p4 = tile1 & 0xFFFFu;                          // tile_dim1 (tile_dim2=0)
  u32 p5 = stride0;                                  // tensor_dim0_stride[31:0]
  tdm_v8i g1 = { (int)p0, (int)p1, (int)p2, (int)p3, (int)p4, (int)p5, 0, 0 };
  tdm_v4i z4 = { 0, 0, 0, 0 };                       // 2D: groups 2/3 unused
  tdm_v8i z8 = { 0, 0, 0, 0, 0, 0, 0, 0 };
  // 6-arg (clang-23 / therock-10.0) form
  __builtin_amdgcn_tensor_load_to_lds(g0, g1, z4, z4, z8, 0);
}
#endif

// -------- kernel 0: fp32 -> bf16 convert (embedding) --------
__global__ __launch_bounds__(256) void cvt_bf16_kernel(
    const float* __restrict__ in, u16* __restrict__ out, int n) {
  int i = (blockIdx.x * 256 + threadIdx.x) * 4;
  if (i + 3 < n) {
    float4 f = *(const float4*)(in + i);
    out[i + 0] = f2bf(f.x); out[i + 1] = f2bf(f.y);
    out[i + 2] = f2bf(f.z); out[i + 3] = f2bf(f.w);
  }
}

// -------- kernel 1: weight transpose + convert: Wt[n][k] = bf16(W[k][n]) ----
__global__ __launch_bounds__(256) void wtr_kernel(
    const float* __restrict__ W, u16* __restrict__ Wt) {
  __shared__ float tile[32][33];
  int tx = threadIdx.x & 31, ty = threadIdx.x >> 5;  // 32x8
  int n0 = blockIdx.x * 32, k0 = blockIdx.y * 32;
  #pragma unroll
  for (int i = 0; i < 32; i += 8)
    tile[ty + i][tx] = W[(size_t)(k0 + ty + i) * D_ + n0 + tx];
  __syncthreads();
  #pragma unroll
  for (int i = 0; i < 32; i += 8)
    Wt[(size_t)(n0 + ty + i) * D_ + k0 + tx] = f2bf(tile[tx][ty + i]);
}

// -------- shared GEMM core: C(BMxBN) = A(bf16 MxK) * Bt(bf16 [N][K])^T -----
#define BM 128
#define BN 64
#define BK 64
#define SA 72   // padded LDS row strides (16B-aligned rows, odd dword pairs)
#define SB 72

__device__ __forceinline__ void gemm_tiles(
    const u16* __restrict__ A, const u16* __restrict__ Bt,
    int K, int Marows, int Nbrows, int gm0, int gn0,
    u16* Atile, u16* Btile, v8f acc[2][2]) {
  const int tid = threadIdx.x;
  const int lane = tid & 31, wid = tid >> 5;
  const int lr = lane & 15, half = lane >> 4;
  const int mw = (wid >> 1) * 32, nw = (wid & 1) * 32;  // wave tile origin

#if HAS_TDM
  const bool issuer = (wid == 0);       // one wave drives the TDM engine
  if (issuer) {
    tdm_load_2d(A  + (size_t)gm0 * K, Atile, K, (u32)Marows, K, BK, BM, 4, 3);
    tdm_load_2d(Bt + (size_t)gn0 * K, Btile, K, (u32)Nbrows, K, BK, BN, 4, 3);
    __builtin_amdgcn_s_wait_tensorcnt(0);
  }
  __syncthreads();
#endif

  for (int k0 = 0; k0 < K; k0 += BK) {
#if HAS_TDM
    const int cur = (k0 >> 6) & 1, nxt = cur ^ 1;
    u16* Acur = Atile + cur * (BM * SA);
    u16* Bcur = Btile + cur * (BN * SB);
    if (issuer && k0 + BK < K) {        // DMA next tiles while we compute
      tdm_load_2d(A  + (size_t)gm0 * K + (k0 + BK), Atile + nxt * (BM * SA),
                  K, (u32)Marows, K, BK, BM, 4, 3);
      tdm_load_2d(Bt + (size_t)gn0 * K + (k0 + BK), Btile + nxt * (BN * SB),
                  K, (u32)Nbrows, K, BK, BN, 4, 3);
    }
#else
    u16* Acur = Atile;
    u16* Bcur = Btile;
    // cooperative stage: A 128x64 bf16 (16KB), Bt 64x64 bf16 (8KB), b128 each
    #pragma unroll
    for (int i = 0; i < 4; i++) {
      int idx = tid + i * 256, row = idx >> 3, seg = idx & 7;
      *(uint4*)&Acur[row * SA + seg * 8] =
          *(const uint4*)&A[(size_t)(gm0 + row) * K + k0 + seg * 8];
    }
    #pragma unroll
    for (int i = 0; i < 2; i++) {
      int idx = tid + i * 256, row = idx >> 3, seg = idx & 7;
      *(uint4*)&Bcur[row * SB + seg * 8] =
          *(const uint4*)&Bt[(size_t)(gn0 + row) * K + k0 + seg * 8];
    }
    __syncthreads();
#endif
    #pragma unroll
    for (int ks = 0; ks < BK; ks += 32) {
      Frag af[2], bfr[2];
      #pragma unroll
      for (int im = 0; im < 2; im++) {           // A 16x32: rows in lanes,
        int row = mw + im * 16 + lr;             // 16 contiguous K per half
        af[im].q[0] = *(const uint4*)&Acur[row * SA + ks + half * 16];
        af[im].q[1] = *(const uint4*)&Acur[row * SA + ks + half * 16 + 8];
      }
      #pragma unroll
      for (int in_ = 0; in_ < 2; in_++) {        // B 32x16: lane = column,
        int col = nw + in_ * 16 + lr;            // 16 contiguous K per half
        bfr[in_].q[0] = *(const uint4*)&Bcur[col * SB + ks + half * 16];
        bfr[in_].q[1] = *(const uint4*)&Bcur[col * SB + ks + half * 16 + 8];
      }
      #pragma unroll
      for (int im = 0; im < 2; im++)
        #pragma unroll
        for (int in_ = 0; in_ < 2; in_++)
          acc[im][in_] = wmma_bf16(af[im].v, bfr[in_].v, acc[im][in_]);
    }
#if HAS_TDM
    if (issuer && k0 + BK < K) __builtin_amdgcn_s_wait_tensorcnt(0);
#endif
    __syncthreads();
  }
}

// -------- kernel 2: fused QKV projection (grid.z selects Q/K/V) --------
__global__ __launch_bounds__(256) void qkv_gemm_kernel(
    const u16* __restrict__ Xbf, const u16* __restrict__ WtAll,
    const float* __restrict__ bq, const float* __restrict__ bk,
    const float* __restrict__ bvv,
    u16* __restrict__ Qbf, u16* __restrict__ Kbf, u16* __restrict__ Vt) {
  __shared__ __align__(16) u16 Atile[2 * BM * SA];
  __shared__ __align__(16) u16 Btile[2 * BN * SB];
  const int sel = blockIdx.z;
  const u16* Bt = WtAll + (size_t)sel * D_ * D_;
  const float* bias = (sel == 0) ? bq : (sel == 1) ? bk : bvv;
  const int gm0 = blockIdx.y * BM, gn0 = blockIdx.x * BN;
  v8f acc[2][2] = {};
  gemm_tiles(Xbf, Bt, D_, MTOT, D_, gm0, gn0, Atile, Btile, acc);

  const int tid = threadIdx.x, lane = tid & 31, wid = tid >> 5;
  const int lr = lane & 15, half = lane >> 4;
  const int mw = (wid >> 1) * 32, nw = (wid & 1) * 32;
  #pragma unroll
  for (int im = 0; im < 2; im++)
    #pragma unroll
    for (int in_ = 0; in_ < 2; in_++) {
      int n = gn0 + nw + in_ * 16 + lr;          // lane owns one output column
      int h = n >> 6, d = n & 63;
      float bias_n = bias[n];
      #pragma unroll
      for (int r = 0; r < 8; r++) {              // C layout: VGPR r -> row
        int m = gm0 + mw + im * 16 + half * 8 + r;
        int b = m >> 11, s = m & (S_ - 1);
        float val = acc[im][in_][r] + bias_n;
        if (sel == 0)      Qbf[(size_t)((b * H_ + h) * S_ + s) * DQ_ + d] = f2bf(val);
        else if (sel == 1) Kbf[(size_t)((b * H_ + h) * S_ + s) * DQ_ + d] = f2bf(val);
        else               Vt [(size_t)((b * H_ + h) * DQ_ + d) * S_ + s] = f2bf(val);
      }
    }
}

// -------- kernel 3: flash attention (online softmax, all-WMMA) --------
__global__ __launch_bounds__(128) void attn_kernel(
    const u16* __restrict__ Qbf, const u16* __restrict__ Kbf,
    const u16* __restrict__ Vt, const int* __restrict__ mask,
    u16* __restrict__ Ctx) {
  __shared__ __align__(16) u16 Plds[4][16 * 40];  // per-wave P staging
#if HAS_TDM
  __shared__ __align__(16) u16 KbufL[2][32 * 72]; // 32 keys x 64 d (pad 72)
  __shared__ __align__(16) u16 VbufL[2][64 * 40]; // 64 d x 32 keys (pad 40)
#endif
  const int bh = blockIdx.y, b = bh >> 4, h = bh & 15;
  const int tid = threadIdx.x, wid = tid >> 5, lane = tid & 31;
  const int lr = lane & 15, half = lane >> 4;
  const int q0 = blockIdx.x * 64 + wid * 16;      // 16 queries per wave
  const u16* Qh = Qbf + (size_t)bh * S_ * DQ_;
  const u16* Kh = Kbf + (size_t)bh * S_ * DQ_;
  const u16* Vh = Vt  + (size_t)bh * DQ_ * S_;
  const int* mrow = mask + b * S_;
  u16* P = &Plds[wid][0];

  Frag qf[2];                                     // Q tile 16x64, kept in VGPRs
  #pragma unroll
  for (int i = 0; i < 2; i++) {
    int d0 = i * 32, row = q0 + lr;
    qf[i].q[0] = *(const uint4*)&Qh[(size_t)row * DQ_ + d0 + half * 16];
    qf[i].q[1] = *(const uint4*)&Qh[(size_t)row * DQ_ + d0 + half * 16 + 8];
  }
  v8f o[4] = {};
  float mrun[8], lrun[8];
  #pragma unroll
  for (int r = 0; r < 8; r++) { mrun[r] = -1e30f; lrun[r] = 0.0f; }

#if HAS_TDM
  const bool issuer = (wid == 0);   // one DMA per block, not 4 redundant waves
  if (issuer) {
    tdm_load_2d(Kh, &KbufL[0][0], DQ_, S_, DQ_, DQ_, 32, 4, 3);
    tdm_load_2d(Vh, &VbufL[0][0], S_, DQ_, S_, 32, DQ_, 3, 3);
    __builtin_amdgcn_s_wait_tensorcnt(0);
  }
  __syncthreads();
#endif

  for (int kc = 0; kc < S_; kc += 32) {           // 32-key chunks
#if HAS_TDM
    const int cur = (kc >> 5) & 1, nxt = cur ^ 1;
    const u16* Kc = &KbufL[cur][0];
    const u16* Vc = &VbufL[cur][0];
    if (issuer && kc + 32 < S_) {                 // prefetch next chunk via TDM
      tdm_load_2d(Kh + (size_t)(kc + 32) * DQ_, &KbufL[nxt][0],
                  DQ_, S_, DQ_, DQ_, 32, 4, 3);
      tdm_load_2d(Vh + (kc + 32), &VbufL[nxt][0],
                  S_, DQ_, S_, 32, DQ_, 3, 3);
    }
#endif
    v8f sc[2] = {};
    #pragma unroll
    for (int nt = 0; nt < 2; nt++) {              // S = Q * K^T  (4 WMMAs)
      #pragma unroll
      for (int i = 0; i < 2; i++) {
        int d0 = i * 32;
        Frag kf;  // B-frag of K^T: lane = key column, contiguous d per half
#if HAS_TDM
        int krow = nt * 16 + lr;                  // key index within chunk
        kf.q[0] = *(const uint4*)&Kc[krow * 72 + d0 + half * 16];
        kf.q[1] = *(const uint4*)&Kc[krow * 72 + d0 + half * 16 + 8];
#else
        int key = kc + nt * 16 + lr;
        kf.q[0] = *(const uint4*)&Kh[(size_t)key * DQ_ + d0 + half * 16];
        kf.q[1] = *(const uint4*)&Kh[(size_t)key * DQ_ + d0 + half * 16 + 8];
#endif
        sc[nt] = wmma_bf16(qf[i].v, kf.v, sc[nt]);
      }
    }
    #pragma unroll
    for (int nt = 0; nt < 2; nt++) {              // scale 1/sqrt(64) + mask
      int key = kc + nt * 16 + lr;                // lane's 8 vals share column
      bool keep = (mrow[key] != 0);
      #pragma unroll
      for (int r = 0; r < 8; r++)
        sc[nt][r] = keep ? sc[nt][r] * 0.125f : -1e9f;
    }
    float alpha[8];
    #pragma unroll
    for (int r = 0; r < 8; r++) {                 // online softmax per row
      float mx = fmaxf(sc[0][r], sc[1][r]);
      #pragma unroll
      for (int off = 8; off >= 1; off >>= 1)      // row lives in a 16-lane half
        mx = fmaxf(mx, __shfl_xor(mx, off, 32));
      float mnew = fmaxf(mrun[r], mx);
      float a  = __expf(mrun[r] - mnew);
      float p0 = __expf(sc[0][r] - mnew);
      float p1 = __expf(sc[1][r] - mnew);
      float rs = p0 + p1;
      #pragma unroll
      for (int off = 8; off >= 1; off >>= 1) rs += __shfl_xor(rs, off, 32);
      lrun[r] = lrun[r] * a + rs;
      mrun[r] = mnew;
      alpha[r] = a;
      P[(half * 8 + r) * 40 + lr]      = f2bf(p0);  // C-layout -> LDS
      P[(half * 8 + r) * 40 + 16 + lr] = f2bf(p1);
    }
    #pragma unroll
    for (int j = 0; j < 4; j++)
      #pragma unroll
      for (int r = 0; r < 8; r++) o[j][r] *= alpha[r];

    asm volatile("" ::: "memory");  // LDS round-trip is wave-local + in-order
    Frag pf;                        // reload P in A-frag layout
    pf.q[0] = *(const uint4*)&P[lr * 40 + half * 16];
    pf.q[1] = *(const uint4*)&P[lr * 40 + half * 16 + 8];
    #pragma unroll
    for (int j = 0; j < 4; j++) {                 // O += P * V  (4 WMMAs)
      int d = j * 16 + lr;
      Frag vf;  // B-frag of V: contiguous keys per lane
#if HAS_TDM
      vf.q[0] = *(const uint4*)&Vc[d * 40 + half * 16];
      vf.q[1] = *(const uint4*)&Vc[d * 40 + half * 16 + 8];
#else
      vf.q[0] = *(const uint4*)&Vh[(size_t)d * S_ + kc + half * 16];
      vf.q[1] = *(const uint4*)&Vh[(size_t)d * S_ + kc + half * 16 + 8];
#endif
      o[j] = wmma_bf16(pf.v, vf.v, o[j]);
    }
    asm volatile("" ::: "memory");
#if HAS_TDM
    if (issuer && kc + 32 < S_) __builtin_amdgcn_s_wait_tensorcnt(0);
    __syncthreads();
#endif
  }
  float inv[8];
  #pragma unroll
  for (int r = 0; r < 8; r++) inv[r] = 1.0f / lrun[r];
  #pragma unroll
  for (int j = 0; j < 4; j++) {
    int col = h * DQ_ + j * 16 + lr;
    #pragma unroll
    for (int r = 0; r < 8; r++) {
      int srow = q0 + half * 8 + r;
      Ctx[(size_t)(b * S_ + srow) * D_ + col] = f2bf(o[j][r] * inv[r]);
    }
  }
}

// -------- kernel 4: output projection, fp32 result --------
__global__ __launch_bounds__(256) void oproj_gemm_kernel(
    const u16* __restrict__ Ctx, const u16* __restrict__ WoT,
    const float* __restrict__ bo, float* __restrict__ out) {
  __shared__ __align__(16) u16 Atile[2 * BM * SA];
  __shared__ __align__(16) u16 Btile[2 * BN * SB];
  const int gm0 = blockIdx.y * BM, gn0 = blockIdx.x * BN;
  v8f acc[2][2] = {};
  gemm_tiles(Ctx, WoT, D_, MTOT, D_, gm0, gn0, Atile, Btile, acc);

  const int tid = threadIdx.x, lane = tid & 31, wid = tid >> 5;
  const int lr = lane & 15, half = lane >> 4;
  const int mw = (wid >> 1) * 32, nw = (wid & 1) * 32;
  #pragma unroll
  for (int im = 0; im < 2; im++)
    #pragma unroll
    for (int in_ = 0; in_ < 2; in_++) {
      int n = gn0 + nw + in_ * 16 + lr;
      float bias_n = bo[n];
      #pragma unroll
      for (int r = 0; r < 8; r++) {
        int m = gm0 + mw + im * 16 + half * 8 + r;
        out[(size_t)m * D_ + n] = acc[im][in_][r] + bias_n;
      }
    }
}

// -------- host side --------
extern "C" void kernel_launch(void* const* d_in, const int* in_sizes, int n_in,
                              void* d_out, int out_size, void* d_ws, size_t ws_size,
                              hipStream_t stream) {
  const float* emb = (const float*)d_in[0];
  const int*  mask = (const int*)d_in[1];
  const float* Wq = (const float*)d_in[2]; const float* bq = (const float*)d_in[3];
  const float* Wk = (const float*)d_in[4]; const float* bk = (const float*)d_in[5];
  const float* Wv = (const float*)d_in[6]; const float* bv = (const float*)d_in[7];
  const float* Wo = (const float*)d_in[8]; const float* bo = (const float*)d_in[9];
  float* out = (float*)d_out;

  // workspace carve-up (all bf16): 48 MB total
  char* w = (char*)d_ws;
  u16* Xbf   = (u16*)w; w += (size_t)MTOT * D_ * 2;        // 8 MB
  u16* WtAll = (u16*)w; w += (size_t)4 * D_ * D_ * 2;      // 8 MB
  u16* Qbf   = (u16*)w; w += (size_t)B_ * H_ * S_ * DQ_ * 2;
  u16* Kbf   = (u16*)w; w += (size_t)B_ * H_ * S_ * DQ_ * 2;
  u16* Vt    = (u16*)w; w += (size_t)B_ * H_ * DQ_ * S_ * 2;
  u16* Ctx   = (u16*)w; w += (size_t)MTOT * D_ * 2;

  cvt_bf16_kernel<<<(MTOT * D_) / 1024, 256, 0, stream>>>(emb, Xbf, MTOT * D_);
  dim3 tg(D_ / 32, D_ / 32);
  wtr_kernel<<<tg, 256, 0, stream>>>(Wq, WtAll + 0 * (size_t)D_ * D_);
  wtr_kernel<<<tg, 256, 0, stream>>>(Wk, WtAll + 1 * (size_t)D_ * D_);
  wtr_kernel<<<tg, 256, 0, stream>>>(Wv, WtAll + 2 * (size_t)D_ * D_);
  wtr_kernel<<<tg, 256, 0, stream>>>(Wo, WtAll + 3 * (size_t)D_ * D_);

  qkv_gemm_kernel<<<dim3(D_ / BN, MTOT / BM, 3), 256, 0, stream>>>(
      Xbf, WtAll, bq, bk, bv, Qbf, Kbf, Vt);
  attn_kernel<<<dim3(S_ / 64, B_ * H_), 128, 0, stream>>>(
      Qbf, Kbf, Vt, mask, Ctx);
  oproj_gemm_kernel<<<dim3(D_ / BN, MTOT / BM), 256, 0, stream>>>(
      Ctx, WtAll + 3 * (size_t)D_ * D_, bo, out);
}
